// RepSetRNN_15960098472410
// MI455X (gfx1250) — matrix operand order repaired and checked by reading, not compiled, not run on hardware
//
#include <hip/hip_runtime.h>
#include <hip/hip_bf16.h>
#include <stddef.h>

typedef __attribute__((ext_vector_type(16))) __bf16 v16bf;
typedef __attribute__((ext_vector_type(8)))  __bf16 v8bf;
typedef __attribute__((ext_vector_type(8)))  float  v8f;

// fp32 -> bf16 round-to-nearest-even (bit-level)
static __device__ inline __bf16 f2bf(float f) {
  unsigned u = __float_as_uint(f);
  unsigned r = (u + 0x7FFFu + ((u >> 16) & 1u)) >> 16;
  unsigned short s = (unsigned short)r;
  __bf16 h;
  __builtin_memcpy(&h, &s, sizeof(h));
  return h;
}

static __device__ inline v8f wmma_bf16(v16bf a, v16bf b, v8f c) {
  return __builtin_amdgcn_wmma_f32_16x16x32_bf16(
      false, a, false, b, (short)0, c, false, false);
}

static __device__ inline float max8(v8f c) {
  return fmaxf(fmaxf(fmaxf(c[0], c[1]), fmaxf(c[2], c[3])),
               fmaxf(fmaxf(c[4], c[5]), fmaxf(c[6], c[7])));
}

// ---- CDNA5 async global->LDS staging (ASYNCcnt) with sync fallback ----
typedef int v4i __attribute__((vector_size(16)));
#if __has_builtin(__builtin_amdgcn_global_load_async_to_lds_b128)
#define HAVE_ASYNC_LDS 1
typedef __attribute__((address_space(1))) v4i* gas_v4i_p;   // diag prints AS1 as __device__
typedef __attribute__((address_space(3))) v4i* las_v4i_p;
#endif

static __device__ inline void wait_async_le4() {
#if __has_builtin(__builtin_amdgcn_s_wait_asynccnt)
  __builtin_amdgcn_s_wait_asynccnt(4);
#else
  asm volatile("s_wait_asynccnt 4" ::: "memory");
#endif
}
static __device__ inline void wait_async_le0() {
#if __has_builtin(__builtin_amdgcn_s_wait_asynccnt)
  __builtin_amdgcn_s_wait_asynccnt(0);
#else
  asm volatile("s_wait_asynccnt 0" ::: "memory");
#endif
}

// Stage one 2KB Xc tile (16 rows x 128B) into LDS with 144B row pitch.
// Called by one wave (32 lanes); 4 async b128 issues cover 128 x 16B.
static __device__ inline void stage_tile(const char* gsrc, char* lbuf, int lane) {
#pragma unroll
  for (int j = 0; j < 4; ++j) {
    int idx = j * 32 + lane;          // 0..127
    int row = idx >> 3, seg = idx & 7;
    const char* g = gsrc + row * 128 + seg * 16;
    char* l = lbuf + row * 144 + seg * 16;
#ifdef HAVE_ASYNC_LDS
    __builtin_amdgcn_global_load_async_to_lds_b128(
        (gas_v4i_p)(size_t)g, (las_v4i_p)(unsigned)(size_t)l, 0, 0);
#else
    *(int4*)l = *(const int4*)g;      // synchronous fallback; barriers order it
#endif
  }
}

// ---------------- Kernel 0: hidden_sets fp32 -> bf16 ----------------
__global__ __launch_bounds__(256)
void cvt_h_bf16_kernel(const float* __restrict__ in, __bf16* __restrict__ out, int n) {
  int i = blockIdx.x * 256 + threadIdx.x;
  if (i < n) out[i] = f2bf(in[i]);
}

// ---------------- Kernel 1: Xc = Xs.W^T + b (bf16 out) ----------------
__global__ __launch_bounds__(256)
void compress_kernel(const float* __restrict__ Xs, const float* __restrict__ W,
                     const float* __restrict__ bias, __bf16* __restrict__ Xc) {
  const int tid  = threadIdx.x;
  const int k    = tid & 63;
  const int rloc = tid >> 6;
  const int row  = blockIdx.x * 4 + rloc;
  const float4* x4 = (const float4*)(Xs + (size_t)row * 300);
  const float4* w4 = (const float4*)(W  + (size_t)k   * 300);
  float acc = 0.f;
#pragma unroll 5
  for (int d = 0; d < 75; ++d) {
    float4 a = x4[d], w = w4[d];
    acc += a.x * w.x + a.y * w.y + a.z * w.z + a.w * w.w;
  }
  Xc[(size_t)row * 64 + k] = f2bf(acc + bias[k]);
}

// ---------------- Kernel 2: fused bf16 WMMA + relu/max(n)/sum(s) ----------------
// G^T tiles: A = H[o] rows (M=n, hoisted), B = Xc^T tile staged once per WG in LDS
// (shared by all 8 waves -> 8x less L2 traffic), double-buffered with async copies.
__global__ __launch_bounds__(256)
void repset_main_kernel(const __bf16* __restrict__ Xc, const __bf16* __restrict__ Hbf,
                        float* __restrict__ out) {
  __shared__ alignas(16) char sbuf[2][16 * 144];   // 2 x 2.25KB (144B pitch, conflict-free)

  const int lane = threadIdx.x & 31;
  const int wave = threadIdx.x >> 5;
  const int b    = blockIdx.y;
  const int o    = blockIdx.x * 8 + wave;
  const int col  = lane & 15;
  const int half = lane >> 4;

  // A fragments (H rows, M = n): 4 M-tiles x 2 K-steps, loaded once from global.
  v16bf Af[4][2];
  const __bf16* hbase = Hbf + (size_t)o * (64 * 64);
#pragma unroll
  for (int t = 0; t < 4; ++t) {
    const __bf16* hrow = hbase + (size_t)(t * 16 + col) * 64;
#pragma unroll
    for (int ks = 0; ks < 2; ++ks) {
      v8bf lo = *(const v8bf*)(hrow + ks * 32 + half * 16);
      v8bf hi = *(const v8bf*)(hrow + ks * 32 + half * 16 + 8);
      v16bf a;
#pragma unroll
      for (int i = 0; i < 8; ++i) { a[i] = lo[i]; a[8 + i] = hi[i]; }
      Af[t][ks] = a;
    }
  }

  const char* xtiles = (const char*)(Xc + (size_t)b * (1024 * 64));  // chunk c at +c*2048

  // Prologue: stage chunks 0 and 1 (8 async b128 outstanding on wave 0).
  if (wave == 0) {
    stage_tile(xtiles + 0 * 2048, sbuf[0], lane);
    stage_tile(xtiles + 1 * 2048, sbuf[1], lane);
  }

  float acc = 0.f;
  for (int chunk = 0; chunk < 64; ++chunk) {
    const int cur = chunk & 1;
    if (wave == 0) {
      if (chunk < 63) wait_async_le4();   // oldest stage (this chunk) complete
      else            wait_async_le0();   // last chunk: only its 4 outstanding
    }
    __syncthreads();                      // data ready for all 8 waves

    // B fragments from LDS: lane (col,half) reads 2 x 32B of row col (144B pitch).
    const char* tile = sbuf[cur];
    v16bf B0 = *(const v16bf*)(tile + col * 144 + half * 32);
    v16bf B1 = *(const v16bf*)(tile + col * 144 + 64 + half * 32);

    v8f C0 = {}, C1 = {}, C2 = {}, C3 = {};
    C0 = wmma_bf16(Af[0][0], B0, C0);
    C1 = wmma_bf16(Af[1][0], B0, C1);
    C2 = wmma_bf16(Af[2][0], B0, C2);
    C3 = wmma_bf16(Af[3][0], B0, C3);
    C0 = wmma_bf16(Af[0][1], B1, C0);
    C1 = wmma_bf16(Af[1][1], B1, C1);
    C2 = wmma_bf16(Af[2][1], B1, C2);
    C3 = wmma_bf16(Af[3][1], B1, C3);

    // max over n (registers), merge half-groups, relu, accumulate over s.
    float m = fmaxf(fmaxf(max8(C0), max8(C1)), fmaxf(max8(C2), max8(C3)));
    m = fmaxf(m, __shfl_xor(m, 16, 32));
    acc += fmaxf(m, 0.f);

    __syncthreads();                      // everyone done reading sbuf[cur]
    if (wave == 0 && chunk + 2 < 64)
      stage_tile(xtiles + (size_t)(chunk + 2) * 2048, sbuf[cur], lane);
  }

  // sum over the 16 s-columns (within one half-group)
#pragma unroll
  for (int off = 1; off < 16; off <<= 1)
    acc += __shfl_xor(acc, off, 32);
  if (lane == 0) out[b * 256 + o] = acc;
}

// --------------------------------------------------------------------------------------
extern "C" void kernel_launch(void* const* d_in, const int* in_sizes, int n_in,
                              void* d_out, int out_size, void* d_ws, size_t ws_size,
                              hipStream_t stream) {
  const float* Xs   = (const float*)d_in[0];   // [16,1024,300]
  const float* W    = (const float*)d_in[1];   // [64,300]
  const float* bias = (const float*)d_in[2];   // [64]
  const float* H    = (const float*)d_in[3];   // [256,64,64]
  float* out = (float*)d_out;                  // [16,256]

  __bf16* Hbf = (__bf16*)d_ws;                               // 2 MB
  __bf16* Xc  = (__bf16*)((char*)d_ws + (size_t)(2u << 20)); // 2 MB

  cvt_h_bf16_kernel<<<4096, 256, 0, stream>>>(H, Hbf, 256 * 64 * 64);
  compress_kernel<<<4096, 256, 0, stream>>>(Xs, W, bias, Xc);
  dim3 grid(32, 16);
  repset_main_kernel<<<grid, 256, 0, stream>>>(Xc, Hbf, out);
}
